// NTLBGRepresentativeSelector_17428977287680
// MI455X (gfx1250) — matrix operand order repaired and checked by reading, not compiled, not run on hardware
//
#include <hip/hip_runtime.h>
#include <hip/hip_bf16.h>
#include <math.h>

#define D     1024
#define HN    8
#define DH    128
#define BB    8
#define TT    4096
#define KSEL  6
#define TEMPR 0.1f
#define LN_EPS 1e-5f

typedef _Float16 half8 __attribute__((ext_vector_type(8)));
typedef _Float16 v16h  __attribute__((ext_vector_type(16)));
typedef float    v8f   __attribute__((ext_vector_type(8)));
typedef float    f4v   __attribute__((ext_vector_type(4)));

// ---------------------------------------------------------------------------
// One 16x16 output tile of  C = A(16xD, f16 in LDS) * W(rows n0..n0+15, f32)^T
// using V_WMMA_F32_16X16X32_F16.  A lane layout per CDNA5 ISA 7.12.2:
//   lanes 0-15  : row = lane,    K in {0..7} U {16..23}  (of each 32-chunk)
//   lanes 16-31 : row = lane-16, K in {8..15} U {24..31}
// B lane layout: lane L holds column n = L%16, K-half selected by L>=16.
// ---------------------------------------------------------------------------
__device__ __forceinline__ v8f wmma_tile_16xD(const _Float16* __restrict__ lA,
                                              const float* __restrict__ Wrows,
                                              int n0)
{
    const int lane = threadIdx.x & 31;
    const int hi   = lane >> 4;        // 0 or 1
    const int l16  = lane & 15;
    const _Float16* ap = lA + l16 * D + hi * 8;
    const float*    wp = Wrows + (size_t)(n0 + l16) * D + hi * 8;
    v8f acc = {};
    for (int k0 = 0; k0 < D; k0 += 32) {
        half8 alo = *(const half8*)(ap + k0);
        half8 ahi = *(const half8*)(ap + k0 + 16);
        v16h A;
#pragma unroll
        for (int i = 0; i < 8; ++i) { A[i] = alo[i]; A[i + 8] = ahi[i]; }
        f4v w0 = *(const f4v*)(wp + k0);
        f4v w1 = *(const f4v*)(wp + k0 + 4);
        f4v w2 = *(const f4v*)(wp + k0 + 16);
        f4v w3 = *(const f4v*)(wp + k0 + 20);
        v16h Bv;
#pragma unroll
        for (int i = 0; i < 4; ++i) {
            Bv[i]      = (_Float16)w0[i];
            Bv[4 + i]  = (_Float16)w1[i];
            Bv[8 + i]  = (_Float16)w2[i];
            Bv[12 + i] = (_Float16)w3[i];
        }
        acc = __builtin_amdgcn_wmma_f32_16x16x32_f16(false, A, false, Bv,
                                                     (short)0, acc, false, false);
    }
    return acc;
}

// ---------------------------------------------------------------------------
// K1: query MLP, one block per branch (0 = mu, 1 = sigma).
//   h = relu(LN(q @ w1.T + b1)); out = h @ w2.T + b2  [softplus+1e-6 if sigma]
// ---------------------------------------------------------------------------
__global__ __launch_bounds__(256)
void k_query_mlp(const float* __restrict__ q,
                 const float* w1m, const float* b1m, const float* gm, const float* bem,
                 const float* w2m, const float* b2m,
                 const float* w1s, const float* b1s, const float* gs, const float* bes,
                 const float* w2s, const float* b2s,
                 float* __restrict__ mu_out, float* __restrict__ sg_out)
{
    const int branch = blockIdx.x;
    const float* w1 = branch ? w1s : w1m;
    const float* b1 = branch ? b1s : b1m;
    const float* g  = branch ? gs  : gm;
    const float* be = branch ? bes : bem;
    const float* w2 = branch ? w2s : w2m;
    const float* b2 = branch ? b2s : b2m;

    __shared__ __align__(16) _Float16 sA[16 * D];   // 32 KB, reused for hidden
    __shared__ float sSum[16], sSq[16], sMean[16], sIstd[16];

    for (int i = threadIdx.x; i < 16 * D; i += 256) {
        int r = i >> 10, c = i & (D - 1);
        sA[i] = (_Float16)((r < BB) ? q[r * D + c] : 0.0f);
    }
    if (threadIdx.x < 16) { sSum[threadIdx.x] = 0.0f; sSq[threadIdx.x] = 0.0f; }
    __syncthreads();

    const int wave = threadIdx.x >> 5;
    const int lane = threadIdx.x & 31;
    const int hi = lane >> 4, l16 = lane & 15;

    v8f acc[8];
#pragma unroll
    for (int t = 0; t < 8; ++t)
        acc[t] = wmma_tile_16xD(sA, w1, (wave * 8 + t) * 16);

#pragma unroll
    for (int t = 0; t < 8; ++t) {
        int col = (wave * 8 + t) * 16 + l16;
        float bb = b1[col];
#pragma unroll
        for (int j = 0; j < 8; ++j) {
            float v = acc[t][j] + bb;
            acc[t][j] = v;
            int row = j + hi * 8;
            atomicAdd(&sSum[row], v);
            atomicAdd(&sSq[row], v * v);
        }
    }
    __syncthreads();
    if (threadIdx.x < 16) {
        float mean = sSum[threadIdx.x] * (1.0f / D);
        float var  = sSq[threadIdx.x] * (1.0f / D) - mean * mean;
        sMean[threadIdx.x] = mean;
        sIstd[threadIdx.x] = rsqrtf(var + LN_EPS);
    }
    __syncthreads();

    // LN + relu -> overwrite sA with hidden (f16)
#pragma unroll
    for (int t = 0; t < 8; ++t) {
        int col = (wave * 8 + t) * 16 + l16;
        float gg = g[col], bbe = be[col];
#pragma unroll
        for (int j = 0; j < 8; ++j) {
            int row = j + hi * 8;
            float v = (acc[t][j] - sMean[row]) * sIstd[row] * gg + bbe;
            sA[row * D + col] = (_Float16)fmaxf(v, 0.0f);
        }
    }
    __syncthreads();

    float* outp = branch ? sg_out : mu_out;
#pragma unroll
    for (int t = 0; t < 8; ++t) {
        int n0 = (wave * 8 + t) * 16;
        v8f a2 = wmma_tile_16xD(sA, w2, n0);
        int col = n0 + l16;
        float bb = b2[col];
#pragma unroll
        for (int j = 0; j < 8; ++j) {
            int row = j + hi * 8;
            if (row < BB) {
                float v = a2[j] + bb;
                if (branch) {
                    v = (v > 20.0f) ? v : log1pf(expf(v));   // softplus
                    v += 1e-6f;
                }
                outp[row * D + col] = v;
            }
        }
    }
}

// ---------------------------------------------------------------------------
// K2: q_proj = mu_q @ wq.T + bq   (wq = in_w rows [0, D))
// ---------------------------------------------------------------------------
__global__ __launch_bounds__(256)
void k_qproj(const float* __restrict__ mu_q, const float* __restrict__ in_w,
             const float* __restrict__ in_b, float* __restrict__ qproj)
{
    __shared__ __align__(16) _Float16 sA[16 * D];
    for (int i = threadIdx.x; i < 16 * D; i += 256) {
        int r = i >> 10, c = i & (D - 1);
        sA[i] = (_Float16)((r < BB) ? mu_q[r * D + c] : 0.0f);
    }
    __syncthreads();
    const int wave = threadIdx.x >> 5, lane = threadIdx.x & 31;
    const int hi = lane >> 4, l16 = lane & 15;
#pragma unroll
    for (int t = 0; t < 8; ++t) {
        int n0 = (wave * 8 + t) * 16;
        v8f a = wmma_tile_16xD(sA, in_w, n0);
        int col = n0 + l16;
        float bb = in_b[col];
#pragma unroll
        for (int j = 0; j < 8; ++j) {
            int row = j + hi * 8;
            if (row < BB) qproj[row * D + col] = a[j] + bb;
        }
    }
}

// ---------------------------------------------------------------------------
// K3: fold query into Wk:  u[b,h,d] = (1/sqrt(dh)) * sum_dh qp[b,h*dh+i]*wk[h*dh+i, d]
//     qc[b,h]             = (1/sqrt(dh)) * qp[b,h,:]·bk_h
// ---------------------------------------------------------------------------
__global__ __launch_bounds__(256)
void k_heads(const float* __restrict__ qproj, const float* __restrict__ in_w,
             const float* __restrict__ in_b, float* __restrict__ u,
             float* __restrict__ qc)
{
    const int bh = blockIdx.x, b = bh >> 3, h = bh & 7;
    __shared__ float sq[DH];
    if (threadIdx.x < DH) sq[threadIdx.x] = qproj[b * D + h * DH + threadIdx.x];
    __syncthreads();
    const float* wk = in_w + (size_t)D * D + (size_t)(h * DH) * D;
    const float rs = 0.08838834764831845f;  // 1/sqrt(128)
    for (int d = threadIdx.x; d < D; d += 256) {
        float s = 0.0f;
        for (int i = 0; i < DH; ++i) s += sq[i] * wk[(size_t)i * D + d];
        u[(size_t)bh * D + d] = s * rs;
    }
    if (threadIdx.x == 0) {
        const float* bk = in_b + D + h * DH;
        float s = 0.0f;
        for (int i = 0; i < DH; ++i) s += sq[i] * bk[i];
        qc[bh] = s * rs;
    }
}

__global__ void k_invsig(const float* __restrict__ sg, float* __restrict__ inv)
{
    int i = blockIdx.x * blockDim.x + threadIdx.x;
    if (i < BB * D) inv[i] = 1.0f / sg[i];
}

// ---------------------------------------------------------------------------
// K4: THE memory-bound pass. One wave per (b,t) row of video_features.
//     dist[b,t] = sum_d (v-mu)^2 * invsig ; scores[b,h,t] = u[b,h]·v + qc[b,h]
// ---------------------------------------------------------------------------
__global__ __launch_bounds__(256)
void k_dist_scores(const float* __restrict__ vf, const float* __restrict__ mu,
                   const float* __restrict__ inv, const float* __restrict__ u,
                   const float* __restrict__ qc,
                   float* __restrict__ dist, float* __restrict__ scores)
{
    const int b = blockIdx.y;
    const int wave = threadIdx.x >> 5, lane = threadIdx.x & 31;
    const int t = blockIdx.x * 8 + wave;
    const float* vrow = vf + ((size_t)b * TT + t) * D;
    const float* mub  = mu + b * D;
    const float* ivb  = inv + b * D;
    const float* ub   = u + (size_t)b * HN * D;
    float dd = 0.0f;
    float s[HN];
#pragma unroll
    for (int h = 0; h < HN; ++h) s[h] = 0.0f;
    for (int d = lane; d < D; d += 32) {
        float v = vrow[d];
        float c = v - mub[d];
        dd = fmaf(c * c, ivb[d], dd);
#pragma unroll
        for (int h = 0; h < HN; ++h) s[h] = fmaf(ub[h * D + d], v, s[h]);
    }
#pragma unroll
    for (int m = 16; m >= 1; m >>= 1) {
        dd += __shfl_xor(dd, m, 32);
#pragma unroll
        for (int h = 0; h < HN; ++h) s[h] += __shfl_xor(s[h], m, 32);
    }
    if (lane == 0) {
        dist[(size_t)b * TT + t] = dd;
#pragma unroll
        for (int h = 0; h < HN; ++h)
            scores[((size_t)b * HN + h) * TT + t] = s[h] + qc[b * HN + h];
    }
}

// ---------------------------------------------------------------------------
// K5: per-batch median (order statistic 2047) via 32-round radix select.
//     dist >= 0 so the f32 bit pattern is order-preserving as u32.
// ---------------------------------------------------------------------------
__global__ __launch_bounds__(256)
void k_median(const float* __restrict__ dist, float* __restrict__ med)
{
    const int b = blockIdx.x;
    const unsigned* dv = (const unsigned*)(dist + (size_t)b * TT);
    __shared__ int sred[256];
    __shared__ unsigned sPrefix;
    __shared__ int sK;
    if (threadIdx.x == 0) { sPrefix = 0u; sK = (TT - 1) / 2; }
    __syncthreads();
    for (int bit = 31; bit >= 0; --bit) {
        unsigned pref = sPrefix;
        unsigned long long ph = (unsigned long long)pref >> (bit + 1);
        int cnt = 0;
        for (int t = threadIdx.x; t < TT; t += 256) {
            unsigned x = dv[t];
            if (((unsigned long long)x >> (bit + 1)) == ph &&
                (((x >> bit) & 1u) == 0u)) cnt++;
        }
        sred[threadIdx.x] = cnt;
        __syncthreads();
        for (int off = 128; off > 0; off >>= 1) {
            if (threadIdx.x < off) sred[threadIdx.x] += sred[threadIdx.x + off];
            __syncthreads();
        }
        if (threadIdx.x == 0) {
            int c0 = sred[0];
            if (sK >= c0) { sK -= c0; sPrefix = pref | (1u << bit); }
        }
        __syncthreads();
    }
    if (threadIdx.x == 0) med[b] = __uint_as_float(sPrefix);
}

__device__ __forceinline__ float block_sum(float v, float* sred)
{
    int tid = threadIdx.x;
    sred[tid] = v; __syncthreads();
    for (int off = 128; off > 0; off >>= 1) {
        if (tid < off) sred[tid] += sred[tid + off];
        __syncthreads();
    }
    float r = sred[0];
    __syncthreads();
    return r;
}

// ---------------------------------------------------------------------------
// K6: combined[b,t] = exp(-|dist-med|/T) * mean_h softmax_t(scores[b,h,:])
// ---------------------------------------------------------------------------
__global__ __launch_bounds__(256)
void k_combined(const float* __restrict__ scores, const float* __restrict__ dist,
                const float* __restrict__ med, float* __restrict__ combined)
{
    const int b = blockIdx.x;
    __shared__ float sred[256];
    __shared__ float mx[HN], sm[HN];
    const float* sb = scores + (size_t)b * HN * TT;
    for (int h = 0; h < HN; ++h) {
        float m = -__builtin_inff();
        for (int t = threadIdx.x; t < TT; t += 256) m = fmaxf(m, sb[h * TT + t]);
        sred[threadIdx.x] = m; __syncthreads();
        for (int off = 128; off > 0; off >>= 1) {
            if (threadIdx.x < off)
                sred[threadIdx.x] = fmaxf(sred[threadIdx.x], sred[threadIdx.x + off]);
            __syncthreads();
        }
        if (threadIdx.x == 0) mx[h] = sred[0];
        __syncthreads();
        float s = 0.0f;
        for (int t = threadIdx.x; t < TT; t += 256) s += expf(sb[h * TT + t] - mx[h]);
        float tot = block_sum(s, sred);
        if (threadIdx.x == 0) sm[h] = tot;
        __syncthreads();
    }
    float mb = med[b];
    for (int t = threadIdx.x; t < TT; t += 256) {
        float a = 0.0f;
#pragma unroll
        for (int h = 0; h < HN; ++h) a += expf(sb[h * TT + t] - mx[h]) / sm[h];
        a *= (1.0f / HN);
        float dw = expf(-fabsf(dist[(size_t)b * TT + t] - mb) / TEMPR);
        combined[(size_t)b * TT + t] = dw * a;
    }
}

__global__ void k_zero(float* __restrict__ a)
{
    if (threadIdx.x < 3) a[threadIdx.x] = 0.0f;
}

// ---------------------------------------------------------------------------
// K7: per-batch greedy selection (K=6), gather rep rows, loss partials.
// ---------------------------------------------------------------------------
__global__ __launch_bounds__(256)
void k_select(const float* __restrict__ combined, const float* __restrict__ dist,
              const float* __restrict__ vf, float* __restrict__ rep_out,
              float* __restrict__ idx_out, float* __restrict__ loss_acc)
{
    const int b = blockIdx.x;
    const int tid = threadIdx.x;
    __shared__ float sc[TT];
    __shared__ float smind[TT];
    __shared__ unsigned ssel[TT / 32];
    __shared__ float svred[256];
    __shared__ int   sired[256];
    __shared__ int   spick[KSEL];
    __shared__ float srd[KSEL];

    for (int t = tid; t < TT; t += 256) sc[t] = combined[(size_t)b * TT + t];
    for (int i = tid; i < TT / 32; i += 256) ssel[i] = 0u;
    __syncthreads();

    // pick 0: plain argmax (first-occurrence tie rule)
    {
        float bv = -__builtin_inff(); int bi = 0x7fffffff;
        for (int t = tid; t < TT; t += 256) {
            float v = sc[t];
            if (v > bv) { bv = v; bi = t; }
        }
        svred[tid] = bv; sired[tid] = bi; __syncthreads();
        for (int off = 128; off > 0; off >>= 1) {
            if (tid < off) {
                float vo = svred[tid + off]; int io = sired[tid + off];
                if (vo > svred[tid] || (vo == svred[tid] && io < sired[tid])) {
                    svred[tid] = vo; sired[tid] = io;
                }
            }
            __syncthreads();
        }
        if (tid == 0) {
            int p = sired[0];
            spick[0] = p;
            ssel[p >> 5] |= 1u << (p & 31);
        }
        __syncthreads();
        int p0 = spick[0];
        for (int t = tid; t < TT; t += 256)
            smind[t] = fabsf((float)(t - p0));
        __syncthreads();
    }

    for (int k = 1; k < KSEL; ++k) {
        float bv = -__builtin_inff(); int bi = 0x7fffffff;
        for (int t = tid; t < TT; t += 256) {
            bool sel = (ssel[t >> 5] >> (t & 31)) & 1u;
            float v = sel ? -__builtin_inff() : smind[t] * sc[t];
            if (v > bv) { bv = v; bi = t; }
        }
        svred[tid] = bv; sired[tid] = bi; __syncthreads();
        for (int off = 128; off > 0; off >>= 1) {
            if (tid < off) {
                float vo = svred[tid + off]; int io = sired[tid + off];
                if (vo > svred[tid] || (vo == svred[tid] && io < sired[tid])) {
                    svred[tid] = vo; sired[tid] = io;
                }
            }
            __syncthreads();
        }
        if (tid == 0) {
            int p = sired[0];
            spick[k] = p;
            ssel[p >> 5] |= 1u << (p & 31);
        }
        __syncthreads();
        int pk = spick[k];
        for (int t = tid; t < TT; t += 256)
            smind[t] = fminf(smind[t], fabsf((float)(t - pk)));
        __syncthreads();
    }

    // outputs: indices (as float) + gathered rep rows
    if (tid < KSEL) {
        idx_out[b * KSEL + tid] = (float)spick[tid];
        srd[tid] = dist[(size_t)b * TT + spick[tid]];
    }
    __syncthreads();
    for (int i = tid; i < KSEL * D; i += 256) {
        int k = i >> 10, d = i & (D - 1);
        rep_out[((size_t)b * KSEL + k) * D + d] =
            vf[((size_t)b * TT + spick[k]) * D + d];
    }

    // ellipsoid + consistency partials
    if (tid == 0) {
        float v[KSEL];
        for (int k = 0; k < KSEL; ++k) v[k] = srd[k];
        for (int i = 1; i < KSEL; ++i) {           // insertion sort (6 elems)
            float x = v[i]; int j = i - 1;
            while (j >= 0 && v[j] > x) { v[j + 1] = v[j]; --j; }
            v[j + 1] = x;
        }
        float target = v[(KSEL - 1) / 2];
        float ell = 0.0f, cons = 0.0f;
        for (int k = 0; k < KSEL; ++k) {
            float dlt = srd[k] - target;
            ell += dlt * dlt;
            cons += srd[k];                        // consistency == mean(rep_dist)
        }
        atomicAdd(&loss_acc[0], ell);
        atomicAdd(&loss_acc[1], cons);
    }
    __syncthreads();

    // diversity: sum over upper-triangular pairs of (rep_k · rep_j)^2
    float divAcc = 0.0f;
    for (int kk = 0; kk < KSEL; ++kk)
        for (int jj = kk + 1; jj < KSEL; ++jj) {
            const float* ra = vf + ((size_t)b * TT + spick[kk]) * D;
            const float* rb = vf + ((size_t)b * TT + spick[jj]) * D;
            float p = 0.0f;
            for (int d = tid; d < D; d += 256) p = fmaf(ra[d], rb[d], p);
            float dot = block_sum(p, svred);
            if (tid == 0) divAcc += dot * dot;
        }
    if (tid == 0) atomicAdd(&loss_acc[2], divAcc);
}

__global__ void k_loss(const float* __restrict__ acc, float* __restrict__ loss)
{
    loss[0] = acc[0] * (1.0f / (BB * KSEL))
            + 0.1f  * acc[1] * (1.0f / (BB * KSEL))
            + 0.05f * acc[2] * (1.0f / (BB * KSEL * KSEL));
}

// ---------------------------------------------------------------------------
extern "C" void kernel_launch(void* const* d_in, const int* in_sizes, int n_in,
                              void* d_out, int out_size, void* d_ws, size_t ws_size,
                              hipStream_t stream)
{
    (void)in_sizes; (void)n_in; (void)out_size; (void)ws_size;
    const float* vf    = (const float*)d_in[0];
    const float* q     = (const float*)d_in[1];
    const float* mu_w1 = (const float*)d_in[2];
    const float* mu_b1 = (const float*)d_in[3];
    const float* mu_g  = (const float*)d_in[4];
    const float* mu_be = (const float*)d_in[5];
    const float* mu_w2 = (const float*)d_in[6];
    const float* mu_b2 = (const float*)d_in[7];
    const float* sg_w1 = (const float*)d_in[8];
    const float* sg_b1 = (const float*)d_in[9];
    const float* sg_g  = (const float*)d_in[10];
    const float* sg_be = (const float*)d_in[11];
    const float* sg_w2 = (const float*)d_in[12];
    const float* sg_b2 = (const float*)d_in[13];
    const float* in_w  = (const float*)d_in[14];
    const float* in_b  = (const float*)d_in[15];

    float* out  = (float*)d_out;          // tuple flattened in return order
    float* rep  = out;                    // 8*6*1024
    float* idxo = out + 49152;            // 48
    float* comb = out + 49200;            // 8*4096
    float* dist = out + 81968;            // 8*4096
    float* mu   = out + 114736;           // 8*1024
    float* sg   = out + 122928;           // 8*1024
    float* loss = out + 131120;           // 1

    float* ws = (float*)d_ws;
    size_t o = 0;
    float* qproj  = ws + o; o += (size_t)BB * D;        // 8K
    float* u      = ws + o; o += (size_t)BB * HN * D;   // 64K
    float* qc     = ws + o; o += BB * HN;               // 64
    float* scores = ws + o; o += (size_t)BB * HN * TT;  // 256K
    float* invsig = ws + o; o += (size_t)BB * D;        // 8K
    float* med    = ws + o; o += BB;                    // 8
    float* lacc   = ws + o; o += 3;

    k_query_mlp<<<2, 256, 0, stream>>>(q, mu_w1, mu_b1, mu_g, mu_be, mu_w2, mu_b2,
                                       sg_w1, sg_b1, sg_g, sg_be, sg_w2, sg_b2,
                                       mu, sg);
    k_invsig<<<(BB * D + 255) / 256, 256, 0, stream>>>(sg, invsig);
    k_qproj<<<1, 256, 0, stream>>>(mu, in_w, in_b, qproj);
    k_heads<<<BB * HN, 256, 0, stream>>>(qproj, in_w, in_b, u, qc);
    k_dist_scores<<<dim3(TT / 8, BB), 256, 0, stream>>>(vf, mu, invsig, u, qc,
                                                        dist, scores);
    k_median<<<BB, 256, 0, stream>>>(dist, med);
    k_combined<<<BB, 256, 0, stream>>>(scores, dist, med, comb);
    k_zero<<<1, 64, 0, stream>>>(lacc);
    k_select<<<BB, 256, 0, stream>>>(comb, dist, vf, rep, idxo, lacc);
    k_loss<<<1, 1, 0, stream>>>(lacc, loss);
}
